// ScaledDotProduct_24824910971446
// MI455X (gfx1250) — matrix operand order repaired and checked by reading, not compiled, not run on hardware
//
#include <hip/hip_runtime.h>
#include <math.h>

// ScaledDotProduct attention, BSH=16, S=4096, D=64, fp32 in/out.
// Flash-attention on v_wmma_f32_16x16x32_f16, f16 staging prepass,
// software-pipelined K fragments (ping-pong, unroll-by-2).

typedef __attribute__((ext_vector_type(16))) _Float16 v16h;
typedef __attribute__((ext_vector_type(8)))  _Float16 v8h;
typedef __attribute__((ext_vector_type(8)))  float    v8f;
typedef __attribute__((ext_vector_type(4)))  int      v4i;

#define BSH_   16
#define SEQ_   4096
#define DIM_   64
#define NBLK_  (SEQ_ / 32)   // 128 key blocks
#define WAVES_PER_BLOCK 8    // 256 threads, one 16-query tile per wave

#define SCALE_LOG2E (0.125f * 1.4426950408889634f)

#define WMMA(a, b, c) \
  __builtin_amdgcn_wmma_f32_16x16x32_f16(false, (a), false, (b), (short)0, (c), false, false)

static __device__ __forceinline__ float xor16(float v) {
  return __shfl_xor(v, 16, 32);
}

// 16x32 f16 fragment half-pair: elems 0..7 = p[0..7], elems 8..15 = p[8..15]
static __device__ __forceinline__ v16h frag2(const _Float16* __restrict__ p) {
  v8h lo = *(const v8h*)(p);
  v8h hi = *(const v8h*)(p + 8);
  v16h a;
#pragma unroll
  for (int j = 0; j < 8; ++j) { a[j] = lo[j]; a[8 + j] = hi[j]; }
  return a;
}

static __device__ __forceinline__ void LoadK(v16h (&k)[4],
                                             const _Float16* __restrict__ kbp,
                                             int ql, int h) {
  const _Float16* kr0 = kbp + (size_t)ql * DIM_ + 16 * h;
  const _Float16* kr1 = kbp + (size_t)(16 + ql) * DIM_ + 16 * h;
  k[0] = frag2(kr0);        // keys 0..15, d 0..31
  k[1] = frag2(kr0 + 32);   // keys 0..15, d 32..63
  k[2] = frag2(kr1);        // keys 16..31, d 0..31
  k[3] = frag2(kr1 + 32);   // keys 16..31, d 32..63
}

// Online softmax update + build B-fragment of P^T (element e -> key e + 16h).
static __device__ __forceinline__ v16h SoftmaxPB(v8f st0, v8f st1, int h,
                                                 float& m_run, float& l_run,
                                                 v8f (&o)[4]) {
  float mx = -1.0e30f;
#pragma unroll
  for (int r = 0; r < 8; ++r) mx = fmaxf(mx, fmaxf(st0[r], st1[r]));
  mx = fmaxf(mx, xor16(mx));
  const float m_new = fmaxf(m_run, mx);
  const float alpha = __builtin_amdgcn_exp2f(m_run - m_new);

  float p0[8], p1[8], lsum = 0.0f;
#pragma unroll
  for (int r = 0; r < 8; ++r) {
    p0[r] = __builtin_amdgcn_exp2f(st0[r] - m_new);
    p1[r] = __builtin_amdgcn_exp2f(st1[r] - m_new);
    lsum += p0[r] + p1[r];
  }
  lsum += xor16(lsum);
  l_run = l_run * alpha + lsum;
  m_run = m_new;

#pragma unroll
  for (int t = 0; t < 4; ++t)
#pragma unroll
    for (int r = 0; r < 8; ++r) o[t][r] *= alpha;

  // Pack to f16 first, then half-swap only the 4 dwords the partner needs.
  v8h own0, own1;
#pragma unroll
  for (int r = 0; r < 8; ++r) {
    own0[r] = (_Float16)p0[r];
    own1[r] = (_Float16)p1[r];
  }
  v4i i0 = __builtin_bit_cast(v4i, own0);
  v4i i1 = __builtin_bit_cast(v4i, own1);
  v4i snd = (h == 0) ? i1 : i0;
  v4i rcv;
#pragma unroll
  for (int j = 0; j < 4; ++j) rcv[j] = __shfl_xor(snd[j], 16, 32);
  v4i lo = (h == 0) ? i0 : rcv;
  v4i hi = (h == 0) ? rcv : i1;
  v8h lh = __builtin_bit_cast(v8h, lo);
  v8h hh = __builtin_bit_cast(v8h, hi);
  v16h pb;
#pragma unroll
  for (int j = 0; j < 8; ++j) { pb[j] = lh[j]; pb[8 + j] = hh[j]; }
  return pb;
}

// One 32-key block: issue next block's K loads, this block's V loads,
// then QK WMMAs (data resident), softmax, PV WMMAs.
static __device__ __forceinline__ void Block(const v16h (&cur)[4], v16h (&nxt)[4],
                                             const _Float16* __restrict__ Kbh,
                                             const _Float16* __restrict__ Vtb,
                                             int kbase, int nbase, int ql, int h,
                                             const v16h (&qb)[2], v8f (&o)[4],
                                             float& m_run, float& l_run) {
  LoadK(nxt, Kbh + (size_t)nbase * DIM_, ql, h);

  v16h av[4];
#pragma unroll
  for (int t = 0; t < 4; ++t)
    av[t] = frag2(Vtb + (size_t)(16 * t + ql) * SEQ_ + kbase + 16 * h);

  v8f st0 = (v8f)(0.0f), st1 = (v8f)(0.0f);
  st0 = WMMA(cur[0], qb[0], st0);
  st0 = WMMA(cur[1], qb[1], st0);
  st1 = WMMA(cur[2], qb[0], st1);
  st1 = WMMA(cur[3], qb[1], st1);

  v16h pb = SoftmaxPB(st0, st1, h, m_run, l_run, o);

#pragma unroll
  for (int t = 0; t < 4; ++t) o[t] = WMMA(av[t], pb, o[t]);
}

// ---------------- prepass A: K (f32) -> Kh (f16), same layout ----------------
__global__ __launch_bounds__(256)
void CvtF32F16(const float* __restrict__ src, _Float16* __restrict__ dst) {
  const size_t i = ((size_t)blockIdx.x * 256 + threadIdx.x) * 8;
  v8f v = *(const v8f*)(src + i);
  v8h o;
#pragma unroll
  for (int j = 0; j < 8; ++j) o[j] = (_Float16)v[j];
  *(v8h*)(dst + i) = o;
}

// ------------- prepass B: V [bh][s][d] f32 -> Vt [bh][d][s] f16 --------------
__global__ __launch_bounds__(256)
void TransV(const float* __restrict__ V, _Float16* __restrict__ Vt) {
  __shared__ _Float16 tile[64][72];
  const int b = blockIdx.y;
  const int sbase = blockIdx.x * 64;
  const int t = threadIdx.x;
  {
    const int sl = t >> 2, c = t & 3;
    const float* src = V + ((size_t)b * SEQ_ + sbase + sl) * DIM_ + 16 * c;
#pragma unroll
    for (int j = 0; j < 16; ++j) tile[sl][16 * c + j] = (_Float16)src[j];
  }
  __syncthreads();
  {
    const int d = t >> 2, sc = t & 3;
    _Float16* dst = Vt + ((size_t)b * DIM_ + d) * SEQ_ + sbase + 16 * sc;
#pragma unroll
    for (int j = 0; j < 16; ++j) dst[j] = tile[16 * sc + j][d];
  }
}

// ------------------------------ main kernel ----------------------------------
__global__ __launch_bounds__(256)
void FlashWmmaF16(const float* __restrict__ Q,
                  const _Float16* __restrict__ Kh,   // [bh][s][d] f16
                  const _Float16* __restrict__ Vt,   // [bh][d][s] f16
                  float* __restrict__ O) {
  const int lane = threadIdx.x & 31;
  const int wave = threadIdx.x >> 5;
  const int h    = lane >> 4;
  const int ql   = lane & 15;

  const int b     = blockIdx.y;
  const int qbase = (blockIdx.x * WAVES_PER_BLOCK + wave) * 16;
  const int q     = qbase + ql;

  const float*    __restrict__ Qb  = Q  + (size_t)b * SEQ_ * DIM_;
  const _Float16* __restrict__ Kbh = Kh + (size_t)b * SEQ_ * DIM_;
  const _Float16* __restrict__ Vtb = Vt + (size_t)b * DIM_ * SEQ_;
  float*          __restrict__ Ob  = O  + (size_t)b * SEQ_ * DIM_;

  // B-fragments of Q^T, scale pre-folded. qb[s] element e -> d = 32s + 16h + e.
  v16h qb[2];
  {
    const float* qrow = Qb + (size_t)q * DIM_;
#pragma unroll
    for (int s = 0; s < 2; ++s) {
      const float* src = qrow + 32 * s + 16 * h;
#pragma unroll
      for (int e = 0; e < 16; ++e) qb[s][e] = (_Float16)(src[e] * SCALE_LOG2E);
    }
  }

  v8f o[4];
#pragma unroll
  for (int t = 0; t < 4; ++t) o[t] = (v8f)(0.0f);
  float m_run = -1.0e30f, l_run = 0.0f;

  // Ping-pong K fragments: block kb computes while kb+1 loads.
  v16h kA[4], kB[4];
  LoadK(kA, Kbh, ql, h);

  for (int kb = 0; kb < NBLK_; kb += 2) {
    const int kb0 = kb * 32;
    const int kb1 = kb0 + 32;
    const int nb1 = (kb + 2 < NBLK_) ? kb1 + 32 : 0;  // clamp final prefetch
    Block(kA, kB, Kbh, Vtb, kb0, kb1, ql, h, qb, o, m_run, l_run);
    Block(kB, kA, Kbh, Vtb, kb1, nb1, ql, h, qb, o, m_run, l_run);
  }

  const float inv = 1.0f / l_run;
  float* orow = Ob + (size_t)q * DIM_;
#pragma unroll
  for (int t = 0; t < 4; ++t) {
    const int d0 = 16 * t + 8 * h;
#pragma unroll
    for (int r = 0; r < 8; ++r) orow[d0 + r] = o[t][r] * inv;
  }
}

// ---------------- fallback: direct f32 loads (small workspace) ---------------
__global__ __launch_bounds__(256)
void FlashWmmaDirect(const float* __restrict__ Q,
                     const float* __restrict__ K,
                     const float* __restrict__ V,
                     float* __restrict__ O) {
  const int lane = threadIdx.x & 31;
  const int wave = threadIdx.x >> 5;
  const int h    = lane >> 4;
  const int ql   = lane & 15;

  const int b     = blockIdx.y;
  const int qbase = (blockIdx.x * WAVES_PER_BLOCK + wave) * 16;
  const int q     = qbase + ql;

  const float* __restrict__ Qb = Q + (size_t)b * SEQ_ * DIM_;
  const float* __restrict__ Kb = K + (size_t)b * SEQ_ * DIM_;
  const float* __restrict__ Vb = V + (size_t)b * SEQ_ * DIM_;
  float*       __restrict__ Ob = O + (size_t)b * SEQ_ * DIM_;

  v16h qb[2];
  {
    const float* qrow = Qb + (size_t)q * DIM_;
#pragma unroll
    for (int s = 0; s < 2; ++s) {
      const float* src = qrow + 32 * s + 16 * h;
#pragma unroll
      for (int e = 0; e < 16; ++e) qb[s][e] = (_Float16)(src[e] * SCALE_LOG2E);
    }
  }

  v8f o[4];
#pragma unroll
  for (int t = 0; t < 4; ++t) o[t] = (v8f)(0.0f);
  float m_run = -1.0e30f, l_run = 0.0f;

  for (int kb = 0; kb < NBLK_; ++kb) {
    const int kbase = kb * 32;
    v8f st[2];
#pragma unroll
    for (int kt = 0; kt < 2; ++kt) {
      st[kt] = (v8f)(0.0f);
      const float* krow = Kb + (size_t)(kbase + 16 * kt + ql) * DIM_;
#pragma unroll
      for (int s = 0; s < 2; ++s) {
        const float* src = krow + 32 * s + 16 * h;
        v16h a;
#pragma unroll
        for (int j = 0; j < 8; ++j) {
          a[j] = (_Float16)src[j];
          a[8 + j] = (_Float16)src[8 + j];
        }
        st[kt] = WMMA(a, qb[s], st[kt]);
      }
    }

    v16h pb = SoftmaxPB(st[0], st[1], h, m_run, l_run, o);

#pragma unroll
    for (int t = 0; t < 4; ++t) {
      const float* vcol = Vb + (size_t)(kbase + 16 * h) * DIM_ + 16 * t + ql;
      v16h av;
#pragma unroll
      for (int j = 0; j < 8; ++j) {
        av[j] = (_Float16)vcol[(size_t)j * DIM_];
        av[8 + j] = (_Float16)vcol[(size_t)(8 + j) * DIM_];
      }
      o[t] = WMMA(av, pb, o[t]);
    }
  }

  const float inv = 1.0f / l_run;
  float* orow = Ob + (size_t)q * DIM_;
#pragma unroll
  for (int t = 0; t < 4; ++t) {
    const int d0 = 16 * t + 8 * h;
#pragma unroll
    for (int r = 0; r < 8; ++r) orow[d0 + r] = o[t][r] * inv;
  }
}

extern "C" void kernel_launch(void* const* d_in, const int* in_sizes, int n_in,
                              void* d_out, int out_size, void* d_ws, size_t ws_size,
                              hipStream_t stream) {
  (void)in_sizes; (void)n_in; (void)out_size;
  const float* Q = (const float*)d_in[0];
  const float* K = (const float*)d_in[1];
  const float* V = (const float*)d_in[2];
  float*       O = (float*)d_out;

  const size_t elems = (size_t)BSH_ * SEQ_ * DIM_;      // 4,194,304
  const size_t need  = 2 * elems * sizeof(_Float16);    // 16 MB

  dim3 grid(SEQ_ / (16 * WAVES_PER_BLOCK), BSH_, 1);    // (32, 16)
  dim3 block(32 * WAVES_PER_BLOCK, 1, 1);               // 256 threads

  if (ws_size >= need) {
    _Float16* Kh = (_Float16*)d_ws;
    _Float16* Vt = Kh + elems;
    CvtF32F16<<<dim3((unsigned)(elems / (256 * 8))), 256, 0, stream>>>(K, Kh);
    TransV<<<dim3(SEQ_ / 64, BSH_), 256, 0, stream>>>(V, Vt);
    FlashWmmaF16<<<grid, block, 0, stream>>>(Q, Kh, Vt, O);
  } else {
    FlashWmmaDirect<<<grid, block, 0, stream>>>(Q, K, V, O);
  }
}
